// Model_16415365005681
// MI455X (gfx1250) — matrix operand order repaired and checked by reading, not compiled, not run on hardware
//
#include <hip/hip_runtime.h>
#include <math.h>

typedef _Float16 f16;
typedef __attribute__((ext_vector_type(16))) _Float16 v16h;
typedef __attribute__((ext_vector_type(8)))  _Float16 v8h;
typedef __attribute__((ext_vector_type(8)))  float    v8f;

// ---------------- problem constants ----------------
#define N_LIG   1024
#define N_PROT  8192
#define N_ALL   9216
#define KNN     48
#define NE      49152      // N_LIG * KNN
#define HID     256
#define NF      6
#define KP_EDGE 544        // 2*HID + 24, padded to mult of 32
#define KP_NODE 512
#define DEPTH   5

__device__ __constant__ float c_freqs[NF] = {
    0.41887902047863906f, 1.6755160819145563f, 6.702064327658225f,
    26.8082573106329f, 107.2330292425316f, 428.93211697012645f};

__device__ __forceinline__ float silu_f(float v) { return v / (1.0f + expf(-v)); }

// ---- WMMA fragment helpers (CDNA5 16-bit layouts, wave32) ----
// A (16xK row-major source): lane(m=l15,khalf): elem e -> k0+16*khalf+e
__device__ __forceinline__ v16h load_a_frag(const f16* rowbase, int koff, int khalf)
{
    return *(const v16h*)(rowbase + koff + 16 * khalf);
}
// B from Bt (N x Kp, contiguous-K): lane(n=l15,khalf): elem e -> k0+16*khalf+e
__device__ __forceinline__ v16h load_b_frag(const f16* colbase, int koff, int khalf)
{
    return *(const v16h*)(colbase + koff + 16 * khalf);
}

// ---------------- weight pack: f32 (L,K,N) row-major -> f16 (L,N,Kp) "Bt", zero-padded K ----------------
__global__ void k_pack_w(const float* __restrict__ src, f16* __restrict__ dst,
                         int K, int N, int Kp)
{
    int l = blockIdx.y;
    const float* s = src + (size_t)l * K * N;
    f16* d = dst + (size_t)l * N * Kp;
    int total = N * Kp;
    for (int idx = blockIdx.x * blockDim.x + threadIdx.x; idx < total;
         idx += gridDim.x * blockDim.x) {
        int n = idx / Kp, k = idx - n * Kp;
        d[idx] = (k < K) ? (f16)s[(size_t)k * N + n] : (f16)0.0f;
    }
}

// ---------------- build x (N_ALL x 3) and feats (N_ALL x 40) ----------------
__global__ void k_build(const float* __restrict__ ppos, const int* __restrict__ ele,
                        const int* __restrict__ aa, const int* __restrict__ bb,
                        const float* __restrict__ xt_pos, const float* __restrict__ xt_feat,
                        const int* __restrict__ t, const float* __restrict__ wt_time,
                        const float* __restrict__ w_ele, const float* __restrict__ w_aa,
                        const float* __restrict__ w_bb,
                        float* __restrict__ x, float* __restrict__ feats)
{
    int i = blockIdx.x * blockDim.x + threadIdx.x;
    if (i >= N_ALL) return;
    if (i < N_LIG) {
        x[i*3+0] = xt_pos[i*3+0]; x[i*3+1] = xt_pos[i*3+1]; x[i*3+2] = xt_pos[i*3+2];
        for (int j = 0; j < 32; ++j) feats[i*40+j] = xt_feat[i*32+j];
        int ti = t[i];
        for (int j = 0; j < 8; ++j) feats[i*40+32+j] = wt_time[ti*8+j];
    } else {
        int p = i - N_LIG;
        x[i*3+0] = ppos[p*3+0]; x[i*3+1] = ppos[p*3+1]; x[i*3+2] = ppos[p*3+2];
        int e = ele[p], a = aa[p], b = bb[p];
        for (int j = 0; j < 16; ++j) feats[i*40+j]     = w_ele[e*16+j];
        for (int j = 0; j < 16; ++j) feats[i*40+16+j]  = w_aa[a*16+j];
        for (int j = 0; j <  8; ++j) feats[i*40+32+j]  = w_bb[b*8+j];
    }
}

// ---------------- KNN: one block per ligand atom; LDS-resident d2 array; 48 argmin rounds ----------------
__global__ void k_knn(const float* __restrict__ x, int* __restrict__ col)
{
    extern __shared__ float smem[];
    float* d2s = smem;                    // N_ALL floats
    float* rv  = smem + N_ALL;            // 256
    int*   ri  = (int*)(rv + 256);        // 256
    int i = blockIdx.x;
    int tid = threadIdx.x;
    float px = x[i*3+0], py = x[i*3+1], pz = x[i*3+2];
    for (int j = tid; j < N_ALL; j += 256) {
        float dx = px - x[j*3+0], dy = py - x[j*3+1], dz = pz - x[j*3+2];
        float d2 = dx*dx + dy*dy + dz*dz;
        d2s[j] = (j == i) ? 3.0e38f : d2;
    }
    __syncthreads();
    for (int r = 0; r < KNN; ++r) {
        float bv = 3.0e38f; int bi = 2147483647;
        for (int j = tid; j < N_ALL; j += 256) {
            float v = d2s[j];
            if (v < bv) { bv = v; bi = j; }
        }
        rv[tid] = bv; ri[tid] = bi;
        __syncthreads();
        for (int s = 128; s > 0; s >>= 1) {
            if (tid < s) {
                if (rv[tid+s] < rv[tid] ||
                    (rv[tid+s] == rv[tid] && ri[tid+s] < ri[tid])) {
                    rv[tid] = rv[tid+s]; ri[tid] = ri[tid+s];
                }
            }
            __syncthreads();
        }
        if (tid == 0) { col[i*KNN + r] = ri[0]; d2s[ri[0]] = 3.0e38f; }
        __syncthreads();
    }
}

// ---------------- emb0: fixed sin embedding of initial distances ----------------
__global__ void k_emb0(const float* __restrict__ x, const int* __restrict__ col,
                       float* __restrict__ emb0)
{
    int e = blockIdx.x * blockDim.x + threadIdx.x;
    if (e >= NE) return;
    int r = e / KNN, c = col[e];
    float dx = x[r*3+0]-x[c*3+0], dy = x[r*3+1]-x[c*3+1], dz = x[r*3+2]-x[c*3+2];
    float d = sqrtf(dx*dx + dy*dy + dz*dz + 1e-8f);
    #pragma unroll
    for (int f = 0; f < NF; ++f) {
        float a = d * c_freqs[f];
        emb0[e*12 + f]      = sinf(a);
        emb0[e*12 + NF + f] = cosf(a);
    }
}

// ---------------- per-layer edge geometry: cdiff (f32) + ea (32 f16, zero-padded) ----------------
__global__ void k_edge_geom(const float* __restrict__ x, const int* __restrict__ col,
                            const float* __restrict__ emb0,
                            float* __restrict__ cdiff, f16* __restrict__ ea32)
{
    int e = blockIdx.x * blockDim.x + threadIdx.x;
    if (e >= NE) return;
    int r = e / KNN, c = col[e];
    float dx = x[r*3+0]-x[c*3+0], dy = x[r*3+1]-x[c*3+1], dz = x[r*3+2]-x[c*3+2];
    float d2 = dx*dx + dy*dy + dz*dz;
    float d = sqrtf(d2 + 1e-8f);
    float inv = 1.0f / (d + 1.0f);
    cdiff[e*3+0] = dx*inv; cdiff[e*3+1] = dy*inv; cdiff[e*3+2] = dz*inv;
    f16* dst = ea32 + (size_t)e * 32;
    #pragma unroll
    for (int f = 0; f < NF; ++f) {
        float a = d * c_freqs[f];
        dst[f]      = (f16)sinf(a);
        dst[NF + f] = (f16)cosf(a);
    }
    for (int j = 0; j < 12; ++j) dst[12 + j] = (f16)emb0[e*12 + j];
    for (int j = 24; j < 32; ++j) dst[j] = (f16)0.0f;   // pad K 536..543
}

// ---------------- epilogue shared by all GEMMs ----------------
__device__ __forceinline__ void gemm_epilogue(v8f c, int m0, int khalf, int ncol, int N,
                                              const float* __restrict__ bias,
                                              float* __restrict__ Cf, f16* __restrict__ Ch,
                                              const float* __restrict__ res, int act)
{
    const float bv = bias[ncol];
    #pragma unroll
    for (int r = 0; r < 8; ++r) {
        int mrow = m0 + r + 8 * khalf;     // C layout: VGPR r holds M = r + 8*(lane>>4)
        float v = c[r] + bv;
        if (act == 1) v = silu_f(v);
        if (res) v += res[(size_t)mrow * N + ncol];
        if (Cf) Cf[(size_t)mrow * N + ncol] = v;
        if (Ch) Ch[(size_t)mrow * N + ncol] = (f16)v;
    }
}

// ---------------- generic WMMA GEMM: A (MxKp f16 row-major) @ Bt -> C, 2 N-tiles/wave ----------------
__global__ void __launch_bounds__(256)
k_gemm_f16(const f16* __restrict__ A, const f16* __restrict__ Bt,
           const float* __restrict__ bias,
           float* __restrict__ Cf, f16* __restrict__ Ch,
           const float* __restrict__ res,
           int M, int N, int Kp, int act)
{
    const int lane  = threadIdx.x & 31;
    const int wave  = threadIdx.x >> 5;
    const int m0    = blockIdx.x * 16;
    const int n0    = wave * 32;           // 8 waves cover N=256
    if (n0 >= N) return;
    const int khalf = lane >> 4;
    const int l15   = lane & 15;
    const f16* arow = A  + (size_t)(m0 + l15) * Kp;
    const f16* bt0  = Bt + (size_t)(n0 + l15) * Kp;
    const f16* bt1  = Bt + (size_t)(n0 + 16 + l15) * Kp;
    v8f c0 = {}, c1 = {};
    for (int k0 = 0; k0 < Kp; k0 += 32) {
        v16h a  = load_a_frag(arow, k0, khalf);
        v16h b0 = load_b_frag(bt0, k0, khalf);
        v16h b1 = load_b_frag(bt1, k0, khalf);
        c0 = __builtin_amdgcn_wmma_f32_16x16x32_f16(false, a, false, b0, (short)0, c0, false, false);
        c1 = __builtin_amdgcn_wmma_f32_16x16x32_f16(false, a, false, b1, (short)0, c1, false, false);
    }
    gemm_epilogue(c0, m0, khalf, n0 + l15,      N, bias, Cf, Ch, res, act);
    gemm_epilogue(c1, m0, khalf, n0 + 16 + l15, N, bias, Cf, Ch, res, act);
}

// ---------------- fused edge GEMM: A row = [h16[row] | h16[col] | ea32] gathered on the fly ----------------
__global__ void __launch_bounds__(256)
k_gemm_edge(const f16* __restrict__ h16, const f16* __restrict__ ea32,
            const int* __restrict__ col, const f16* __restrict__ Bt,
            const float* __restrict__ bias,
            float* __restrict__ Cf, f16* __restrict__ Ch, int act)
{
    const int lane  = threadIdx.x & 31;
    const int wave  = threadIdx.x >> 5;
    const int m0    = blockIdx.x * 16;
    const int n0    = wave * 32;
    const int khalf = lane >> 4;
    const int l15   = lane & 15;
    const int e     = m0 + l15;            // this lane's edge (A row)
    const f16* rowb = h16  + (size_t)(e / KNN) * HID;
    const f16* colb = h16  + (size_t)col[e] * HID;
    const f16* eab  = ea32 + (size_t)e * 32;
    const f16* bt0  = Bt + (size_t)(n0 + l15) * KP_EDGE;
    const f16* bt1  = Bt + (size_t)(n0 + 16 + l15) * KP_EDGE;
    v8f c0 = {}, c1 = {};
    #pragma unroll
    for (int ks = 0; ks < HID; ks += 32) {          // K 0..255: h[row]
        v16h a  = load_a_frag(rowb, ks, khalf);
        v16h b0 = load_b_frag(bt0, ks, khalf);
        v16h b1 = load_b_frag(bt1, ks, khalf);
        c0 = __builtin_amdgcn_wmma_f32_16x16x32_f16(false, a, false, b0, (short)0, c0, false, false);
        c1 = __builtin_amdgcn_wmma_f32_16x16x32_f16(false, a, false, b1, (short)0, c1, false, false);
    }
    #pragma unroll
    for (int ks = 0; ks < HID; ks += 32) {          // K 256..511: h[col]
        v16h a  = load_a_frag(colb, ks, khalf);
        v16h b0 = load_b_frag(bt0, HID + ks, khalf);
        v16h b1 = load_b_frag(bt1, HID + ks, khalf);
        c0 = __builtin_amdgcn_wmma_f32_16x16x32_f16(false, a, false, b0, (short)0, c0, false, false);
        c1 = __builtin_amdgcn_wmma_f32_16x16x32_f16(false, a, false, b1, (short)0, c1, false, false);
    }
    {                                               // K 512..543: ea (zero-padded to 32)
        v16h a  = load_a_frag(eab, 0, khalf);
        v16h b0 = load_b_frag(bt0, 2*HID, khalf);
        v16h b1 = load_b_frag(bt1, 2*HID, khalf);
        c0 = __builtin_amdgcn_wmma_f32_16x16x32_f16(false, a, false, b0, (short)0, c0, false, false);
        c1 = __builtin_amdgcn_wmma_f32_16x16x32_f16(false, a, false, b1, (short)0, c1, false, false);
    }
    gemm_epilogue(c0, m0, khalf, n0 + l15,      HID, bias, Cf, Ch, nullptr, act);
    gemm_epilogue(c1, m0, khalf, n0 + 16 + l15, HID, bias, Cf, Ch, nullptr, act);
}

// ---------------- cat2 GEMM: A row = [A0[i] | A1[i]] (each HID wide), Kp = 512 ----------------
__global__ void __launch_bounds__(256)
k_gemm_cat2(const f16* __restrict__ A0, const f16* __restrict__ A1,
            const f16* __restrict__ Bt, const float* __restrict__ bias,
            float* __restrict__ Cf, f16* __restrict__ Ch, int act)
{
    const int lane  = threadIdx.x & 31;
    const int wave  = threadIdx.x >> 5;
    const int m0    = blockIdx.x * 16;
    const int n0    = wave * 32;
    const int khalf = lane >> 4;
    const int l15   = lane & 15;
    const f16* r0   = A0 + (size_t)(m0 + l15) * HID;
    const f16* r1   = A1 + (size_t)(m0 + l15) * HID;
    const f16* bt0  = Bt + (size_t)(n0 + l15) * KP_NODE;
    const f16* bt1  = Bt + (size_t)(n0 + 16 + l15) * KP_NODE;
    v8f c0 = {}, c1 = {};
    #pragma unroll
    for (int ks = 0; ks < HID; ks += 32) {
        v16h a  = load_a_frag(r0, ks, khalf);
        v16h b0 = load_b_frag(bt0, ks, khalf);
        v16h b1 = load_b_frag(bt1, ks, khalf);
        c0 = __builtin_amdgcn_wmma_f32_16x16x32_f16(false, a, false, b0, (short)0, c0, false, false);
        c1 = __builtin_amdgcn_wmma_f32_16x16x32_f16(false, a, false, b1, (short)0, c1, false, false);
    }
    #pragma unroll
    for (int ks = 0; ks < HID; ks += 32) {
        v16h a  = load_a_frag(r1, ks, khalf);
        v16h b0 = load_b_frag(bt0, HID + ks, khalf);
        v16h b1 = load_b_frag(bt1, HID + ks, khalf);
        c0 = __builtin_amdgcn_wmma_f32_16x16x32_f16(false, a, false, b0, (short)0, c0, false, false);
        c1 = __builtin_amdgcn_wmma_f32_16x16x32_f16(false, a, false, b1, (short)0, c1, false, false);
    }
    gemm_epilogue(c0, m0, khalf, n0 + l15,      HID, bias, Cf, Ch, nullptr, act);
    gemm_epilogue(c1, m0, khalf, n0 + 16 + l15, HID, bias, Cf, Ch, nullptr, act);
}

// ---------------- wave-per-edge 256-dot: attention gate or coord scalar ----------------
__global__ void k_edge_dot(const float* __restrict__ m2, const float* __restrict__ w,
                           const float* __restrict__ b, float* __restrict__ outv,
                           int do_sig)
{
    int lane = threadIdx.x & 31;
    int wv   = threadIdx.x >> 5;
    int e    = blockIdx.x * 8 + wv;
    const float* row = m2 + (size_t)e * HID + lane * 8;
    float p = 0.0f;
    #pragma unroll
    for (int q = 0; q < 8; ++q) p += row[q] * w[lane * 8 + q];
    for (int off = 16; off > 0; off >>= 1) p += __shfl_down(p, off, 32);
    if (lane == 0) {
        if (do_sig) outv[e] = 1.0f / (1.0f + expf(-(p + b[0])));
        else        outv[e] = p;
    }
}

// ---------------- gated contiguous aggregation: agg16 (N_ALL x 256 f16) ----------------
__global__ void k_node_agg(const float* __restrict__ m2, const float* __restrict__ gate,
                           f16* __restrict__ agg16)
{
    int i = blockIdx.x;
    int c = threadIdx.x;
    float a = 0.0f;
    if (i < N_LIG) {
        int base = i * KNN;
        for (int q = 0; q < KNN; ++q) {
            int e = base + q;
            a += gate[e] * m2[(size_t)e * HID + c];
        }
        a *= 0.2f;  // / NORM
    }
    agg16[(size_t)i * HID + c] = (f16)a;
}

// ---------------- x update: contiguous scatter per ligand node ----------------
__global__ void k_x_update(float* __restrict__ x, const float* __restrict__ cdiff,
                           const float* __restrict__ s)
{
    int i = blockIdx.x * blockDim.x + threadIdx.x;
    if (i >= N_LIG) return;
    float ax = 0, ay = 0, az = 0;
    int base = i * KNN;
    for (int q = 0; q < KNN; ++q) {
        int e = base + q; float sv = s[e];
        ax += cdiff[e*3+0] * sv; ay += cdiff[e*3+1] * sv; az += cdiff[e*3+2] * sv;
    }
    x[i*3+0] += ax * 0.2f; x[i*3+1] += ay * 0.2f; x[i*3+2] += az * 0.2f;
}

// ---------------- h = feats @ W_in + b_in ----------------
__global__ void k_h_in(const float* __restrict__ feats, const float* __restrict__ Win,
                       const float* __restrict__ bin, float* __restrict__ h,
                       f16* __restrict__ h16)
{
    __shared__ float fs[40];
    int i = blockIdx.x, t = threadIdx.x;
    if (t < 40) fs[t] = feats[i*40 + t];
    __syncthreads();
    float s = bin[t];
    for (int k = 0; k < 40; ++k) s += fs[k] * Win[k*HID + t];
    h  [(size_t)i*HID + t] = s;
    h16[(size_t)i*HID + t] = (f16)s;
}

// ---------------- output head: h[:1024] @ W_out + b_out, and x[:1024]*pos_w ----------------
__global__ void k_out(const float* __restrict__ h, const float* __restrict__ Wout,
                      const float* __restrict__ bout, const float* __restrict__ x,
                      const float* __restrict__ posw, float* __restrict__ out)
{
    int i = blockIdx.x, t = threadIdx.x;
    if (t < 32) {
        float s = bout[t];
        for (int k = 0; k < HID; ++k) s += h[(size_t)i*HID + k] * Wout[k*32 + t];
        out[N_LIG*3 + i*32 + t] = s;
    } else if (t < 35) {
        int d = t - 32;
        out[i*3 + d] = x[i*3 + d] * posw[0];
    }
}

// ---------------- workspace layout ----------------
static constexpr size_t AL(size_t v) { return (v + 255) & ~(size_t)255; }
static constexpr size_t OFF_X    = 0;
static constexpr size_t OFF_COL  = AL(OFF_X    + (size_t)N_ALL*3*4);
static constexpr size_t OFF_EMB0 = AL(OFF_COL  + (size_t)NE*4);
static constexpr size_t OFF_CD   = AL(OFF_EMB0 + (size_t)NE*12*4);
static constexpr size_t OFF_EA   = AL(OFF_CD   + (size_t)NE*3*4);
static constexpr size_t OFF_H    = AL(OFF_EA   + (size_t)NE*32*2);
static constexpr size_t OFF_H16  = AL(OFF_H    + (size_t)N_ALL*HID*4);
static constexpr size_t OFF_M16  = AL(OFF_H16  + (size_t)N_ALL*HID*2);
static constexpr size_t OFF_M2F  = AL(OFF_M16  + (size_t)NE*HID*2);
static constexpr size_t OFF_GATE = AL(OFF_M2F  + (size_t)NE*HID*4);
static constexpr size_t OFF_S    = AL(OFF_GATE + (size_t)NE*4);
static constexpr size_t OFF_AGG  = AL(OFF_S    + (size_t)NE*4);
static constexpr size_t OFF_U16  = AL(OFF_AGG  + (size_t)N_ALL*HID*2);
static constexpr size_t OFF_FEAT = AL(OFF_U16  + (size_t)N_ALL*HID*2);
static constexpr size_t OFF_EW1T = AL(OFF_FEAT + (size_t)N_ALL*40*4);
static constexpr size_t OFF_EW2T = AL(OFF_EW1T + (size_t)DEPTH*HID*KP_EDGE*2);
static constexpr size_t OFF_NW1T = AL(OFF_EW2T + (size_t)DEPTH*HID*HID*2);
static constexpr size_t OFF_NW2T = AL(OFF_NW1T + (size_t)DEPTH*HID*KP_NODE*2);
static constexpr size_t OFF_CW1T = AL(OFF_NW2T + (size_t)DEPTH*HID*HID*2);
static constexpr size_t OFF_CW2T = AL(OFF_CW1T + (size_t)DEPTH*HID*KP_EDGE*2);

extern "C" void kernel_launch(void* const* d_in, const int* in_sizes, int n_in,
                              void* d_out, int out_size, void* d_ws, size_t ws_size,
                              hipStream_t stream)
{
    (void)in_sizes; (void)n_in; (void)out_size; (void)ws_size;
    auto F = [&](int i) { return (const float*)d_in[i]; };
    auto I = [&](int i) { return (const int*)d_in[i]; };
    char* ws = (char*)d_ws;

    float* X     = (float*)(ws + OFF_X);
    int*   COL   = (int*)  (ws + OFF_COL);
    float* EMB0  = (float*)(ws + OFF_EMB0);
    float* CD    = (float*)(ws + OFF_CD);
    f16*   EA    = (f16*)  (ws + OFF_EA);
    float* H     = (float*)(ws + OFF_H);
    f16*   H16   = (f16*)  (ws + OFF_H16);
    f16*   M16   = (f16*)  (ws + OFF_M16);
    float* M2F   = (float*)(ws + OFF_M2F);
    float* GATE  = (float*)(ws + OFF_GATE);
    float* SBUF  = (float*)(ws + OFF_S);
    f16*   AGG16 = (f16*)  (ws + OFF_AGG);
    f16*   U16   = (f16*)  (ws + OFF_U16);
    float* FEAT  = (float*)(ws + OFF_FEAT);
    f16*   EW1T  = (f16*)  (ws + OFF_EW1T);
    f16*   EW2T  = (f16*)  (ws + OFF_EW2T);
    f16*   NW1T  = (f16*)  (ws + OFF_NW1T);
    f16*   NW2T  = (f16*)  (ws + OFF_NW2T);
    f16*   CW1T  = (f16*)  (ws + OFF_CW1T);
    f16*   CW2T  = (f16*)  (ws + OFF_CW2T);

    // weight packing (deterministic each launch)
    k_pack_w<<<dim3((HID*KP_EDGE+255)/256, DEPTH), 256, 0, stream>>>(F(15), EW1T, 536, HID, KP_EDGE);
    k_pack_w<<<dim3((HID*HID+255)/256,     DEPTH), 256, 0, stream>>>(F(17), EW2T, HID, HID, HID);
    k_pack_w<<<dim3((HID*KP_NODE+255)/256, DEPTH), 256, 0, stream>>>(F(21), NW1T, KP_NODE, HID, KP_NODE);
    k_pack_w<<<dim3((HID*HID+255)/256,     DEPTH), 256, 0, stream>>>(F(23), NW2T, HID, HID, HID);
    k_pack_w<<<dim3((HID*KP_EDGE+255)/256, DEPTH), 256, 0, stream>>>(F(25), CW1T, 536, HID, KP_EDGE);
    k_pack_w<<<dim3((HID*HID+255)/256,     DEPTH), 256, 0, stream>>>(F(27), CW2T, HID, HID, HID);

    // inputs -> x, feats
    k_build<<<(N_ALL+255)/256, 256, 0, stream>>>(F(0), I(1), I(2), I(3), F(4), F(5),
                                                 I(6), F(7), F(8), F(9), F(10), X, FEAT);
    // KNN edges
    k_knn<<<N_LIG, 256, (N_ALL + 512) * 4, stream>>>(X, COL);
    // fixed distance embedding
    k_emb0<<<(NE+255)/256, 256, 0, stream>>>(X, COL, EMB0);
    // h = feats @ W_in + b_in
    k_h_in<<<N_ALL, HID, 0, stream>>>(FEAT, F(11), F(12), H, H16);

    for (int l = 0; l < DEPTH; ++l) {
        // geometry (uses current x)
        k_edge_geom<<<NE/256, 256, 0, stream>>>(X, COL, EMB0, CD, EA);
        // edge MLP (gather fused into GEMM)
        k_gemm_edge<<<NE/16, 256, 0, stream>>>(H16, EA, COL, EW1T + (size_t)l*HID*KP_EDGE,
                                               F(16) + l*HID, nullptr, M16, 1);
        k_gemm_f16<<<NE/16, 256, 0, stream>>>(M16, EW2T + (size_t)l*HID*HID,
                                              F(18) + l*HID, M2F, nullptr, nullptr,
                                              NE, HID, HID, 1);
        // attention gate + gated contiguous aggregation
        k_edge_dot<<<NE/8, 256, 0, stream>>>(M2F, F(19) + l*HID, F(20) + l, GATE, 1);
        k_node_agg<<<N_ALL, HID, 0, stream>>>(M2F, GATE, AGG16);
        // node MLP with residual into h
        k_gemm_cat2<<<N_ALL/16, 256, 0, stream>>>(H16, AGG16, NW1T + (size_t)l*HID*KP_NODE,
                                                  F(22) + l*HID, nullptr, U16, 1);
        k_gemm_f16<<<N_ALL/16, 256, 0, stream>>>(U16, NW2T + (size_t)l*HID*HID,
                                                 F(24) + l*HID, H, H16, H,
                                                 N_ALL, HID, HID, 0);
        // coord MLP (uses post-update h via H16, gather fused)
        k_gemm_edge<<<NE/16, 256, 0, stream>>>(H16, EA, COL, CW1T + (size_t)l*HID*KP_EDGE,
                                               F(26) + l*HID, nullptr, M16, 1);
        k_gemm_f16<<<NE/16, 256, 0, stream>>>(M16, CW2T + (size_t)l*HID*HID,
                                              F(28) + l*HID, M2F, nullptr, nullptr,
                                              NE, HID, HID, 1);
        // per-edge coord scalar and x update
        k_edge_dot<<<NE/8, 256, 0, stream>>>(M2F, F(29) + l*HID, F(29), SBUF, 0);
        k_x_update<<<N_LIG/256, 256, 0, stream>>>(X, CD, SBUF);
    }

    // output head
    k_out<<<N_LIG, 64, 0, stream>>>(H, F(13), F(14), X, F(30), (float*)d_out);
}
